// FastSNN_17059610099999
// MI455X (gfx1250) — matrix operand order repaired and checked by reading, not compiled, not run on hardware
//
#include <hip/hip_runtime.h>

// ---- problem constants (from reference) ----
#define T_STEPS 500
#define BATCH   128
#define WIN     256
#define NFILT   512
#define NCLS    10
#define TAU_MEM   0.9f
#define TAU_TRACE 0.95f

typedef __attribute__((ext_vector_type(16))) __bf16 v16bf;
typedef __attribute__((ext_vector_type(8)))  float  v8f;

union Frag {
  v16bf v;
  uint4 q[2];
};

// two f32 -> packed bf16 dword (lo = a). Round-to-nearest (ties away) via
// +0x8000, then one V_PERM_B32 grabs the two high halves. 3 VALU total.
__device__ __forceinline__ unsigned pack2_bf16(float a, float b) {
  unsigned ua = __float_as_uint(a) + 0x8000u;
  unsigned ub = __float_as_uint(b) + 0x8000u;
  // perm(src0, src1, sel): src0 -> byte idx 4..7, src1 -> byte idx 0..3
  return __builtin_amdgcn_perm(ub, ua, 0x07060302u);
}

// Fused: proj GEMM (bf16 WMMA) + LIF scan, spike counts -> counts_out [BATCH][NFILT]
__global__ __launch_bounds__(256) void snn_fused(const float* __restrict__ xw,
                                                 const float* __restrict__ wts,
                                                 float* __restrict__ counts_out) {
  // double-buffered bf16 staging of x[t] slab: 16 rows x 256 cols
  __shared__ __align__(16) unsigned short sm[2][16 * WIN];

  const int tid  = threadIdx.x;
  const int wave = tid >> 5;
  const int lane = tid & 31;
  const int nl   = lane & 15;   // N column (B/C matrices) == M row (A matrix)
  const int h    = lane >> 4;   // half-wave selector
  const int b0   = blockIdx.x * 16;
  const int f0   = blockIdx.y * 128 + wave * 16;

  // ---- preload time-invariant B fragments: B[k][n] = weights[f0+n][k], K=256 ----
  // 32x16 bf16 B tile: lanes 0-15 hold K=0..15, lanes 16-31 hold K=16..31
  Frag bf[8];
  {
    const float* wrow = wts + (size_t)(f0 + nl) * WIN;
#pragma unroll
    for (int kk = 0; kk < 8; ++kk) {
      const float4* p = (const float4*)(wrow + kk * 32 + h * 16);
      float4 w0 = p[0], w1 = p[1], w2 = p[2], w3 = p[3];
      bf[kk].q[0].x = pack2_bf16(w0.x, w0.y);
      bf[kk].q[0].y = pack2_bf16(w0.z, w0.w);
      bf[kk].q[0].z = pack2_bf16(w1.x, w1.y);
      bf[kk].q[0].w = pack2_bf16(w1.z, w1.w);
      bf[kk].q[1].x = pack2_bf16(w2.x, w2.y);
      bf[kk].q[1].y = pack2_bf16(w2.z, w2.w);
      bf[kk].q[1].z = pack2_bf16(w3.x, w3.y);
      bf[kk].q[1].w = pack2_bf16(w3.z, w3.w);
    }
  }

  v8f u   = {0.f,0.f,0.f,0.f,0.f,0.f,0.f,0.f};
  v8f tr  = {0.f,0.f,0.f,0.f,0.f,0.f,0.f,0.f};
  v8f cnt = {0.f,0.f,0.f,0.f,0.f,0.f,0.f,0.f};

  // ---- prologue: slab t=0 -> LDS buf0; slab t=1 -> stA; slab t=2 -> stB ----
  {
    const float* slab = xw + (size_t)b0 * WIN;  // t = 0 (slab = 16*256 contiguous floats)
#pragma unroll
    for (int rep = 0; rep < 4; ++rep) {
      const int d = rep * 1024 + tid * 4;
      float4 vx = *(const float4*)(slab + d);
      uint2 pk;
      pk.x = pack2_bf16(vx.x, vx.y);
      pk.y = pack2_bf16(vx.z, vx.w);
      *(uint2*)&sm[0][d] = pk;
    }
  }
  float4 stA[4], stB[4];
  {
    const float* slab1 = xw + (size_t)1 * (BATCH * WIN) + (size_t)b0 * WIN;
    const float* slab2 = xw + (size_t)2 * (BATCH * WIN) + (size_t)b0 * WIN;
#pragma unroll
    for (int rep = 0; rep < 4; ++rep)
      stA[rep] = *(const float4*)(slab1 + rep * 1024 + tid * 4);
#pragma unroll
    for (int rep = 0; rep < 4; ++rep)
      stB[rep] = *(const float4*)(slab2 + rep * 1024 + tid * 4);
  }
  __syncthreads();

  // One pipeline stage; st holds slab t+1, loaded two iterations ago.
  auto body = [&](int t, int cur, float4 (&st)[4]) {
    // ---- stage slab t+1 from regs into the other LDS buffer ----
    if (t + 1 < T_STEPS) {
      const int nxt = cur ^ 1;
#pragma unroll
      for (int rep = 0; rep < 4; ++rep) {
        const int d = rep * 1024 + tid * 4;
        uint2 pk;
        pk.x = pack2_bf16(st[rep].x, st[rep].y);
        pk.y = pack2_bf16(st[rep].z, st[rep].w);
        *(uint2*)&sm[nxt][d] = pk;
      }
    }

    // ---- issue loads for slab t+3 NOW: two full iterations to complete ----
    if (t + 3 < T_STEPS) {
      const float* slab = xw + (size_t)(t + 3) * (BATCH * WIN) + (size_t)b0 * WIN;
#pragma unroll
      for (int rep = 0; rep < 4; ++rep)
        st[rep] = *(const float4*)(slab + rep * 1024 + tid * 4);
    }

    // ---- proj tile: 8 x WMMA bf16 (K=32 each), two accumulators for ILP ----
    v8f p0 = {0.f,0.f,0.f,0.f,0.f,0.f,0.f,0.f};
    v8f p1 = {0.f,0.f,0.f,0.f,0.f,0.f,0.f,0.f};
    const unsigned short* arow = &sm[cur][nl * WIN];
#pragma unroll
    for (int kk = 0; kk < 8; kk += 2) {
      Frag a0, a1;
      // 16-bit A 16x32 layout (matches B): half h: elems0-15 = K=16h..16h+15
      a0.q[0] = *(const uint4*)(arow + kk * 32 + 16 * h);
      a0.q[1] = *(const uint4*)(arow + kk * 32 + 16 * h + 8);
      a1.q[0] = *(const uint4*)(arow + (kk + 1) * 32 + 16 * h);
      a1.q[1] = *(const uint4*)(arow + (kk + 1) * 32 + 16 * h + 8);
      p0 = __builtin_amdgcn_wmma_f32_16x16x32_bf16(false, a0.v, false, bf[kk].v,
                                                   (short)0, p0, false, false);
      p1 = __builtin_amdgcn_wmma_f32_16x16x32_bf16(false, a1.v, false, bf[kk + 1].v,
                                                   (short)0, p1, false, false);
    }

    // ---- LIF scan update on the 16x16 tile held in 8 VGPRs/lane ----
#pragma unroll
    for (int r = 0; r < 8; ++r) {
      float p   = p0[r] + p1[r];
      float trv = TAU_TRACE * tr[r] + p;
      float uv  = TAU_MEM * u[r] + trv;
      bool  s   = uv > 1.0f;
      cnt[r] += s ? 1.0f : 0.0f;
      u[r]    = s ? 0.0f : uv;
      tr[r]   = trv;
    }

    __syncthreads();
  };

  // unroll by 2 so the stA/stB ping-pong stays in registers (T_STEPS is even)
  for (int t = 0; t < T_STEPS; t += 2) {
    body(t, 0, stA);
    body(t + 1, 1, stB);
  }

  // ---- spike counts out: C-layout lane(nl,h), VGPR r -> row M = r + 8h ----
#pragma unroll
  for (int r = 0; r < 8; ++r) {
    const int b = b0 + r + 8 * h;
    counts_out[(size_t)b * NFILT + f0 + nl] = cnt[r];
  }
}

// decoder: out[b,c] = sum_f counts[b,f] * dec_w[c,f] + dec_b[c]   (deterministic, no atomics)
__global__ __launch_bounds__(256) void snn_decoder(const float* __restrict__ counts,
                                                   const float* __restrict__ dec_w,
                                                   const float* __restrict__ dec_b,
                                                   float* __restrict__ out) {
  const int idx = blockIdx.x * blockDim.x + threadIdx.x;
  if (idx >= BATCH * NCLS) return;
  const int b = idx / NCLS;
  const int c = idx % NCLS;
  const float* cr = counts + (size_t)b * NFILT;
  const float* wr = dec_w + (size_t)c * NFILT;
  float s = dec_b[c];
#pragma unroll 8
  for (int f = 0; f < NFILT; ++f) s = fmaf(cr[f], wr[f], s);
  out[idx] = s;
}

extern "C" void kernel_launch(void* const* d_in, const int* in_sizes, int n_in,
                              void* d_out, int out_size, void* d_ws, size_t ws_size,
                              hipStream_t stream) {
  const float* xw  = (const float*)d_in[0];  // [T,B,W] f32
  const float* wts = (const float*)d_in[1];  // [F,W]   f32
  const float* dw  = (const float*)d_in[2];  // [C,F]   f32
  const float* db  = (const float*)d_in[3];  // [C]     f32
  float* out    = (float*)d_out;             // [B,C]   f32
  float* counts = (float*)d_ws;              // scratch: [B,F] f32 = 256 KB

  dim3 grid(BATCH / 16, NFILT / 128);        // 8 x 4 blocks, 8 waves each
  snn_fused<<<grid, 256, 0, stream>>>(xw, wts, counts);
  snn_decoder<<<(BATCH * NCLS + 255) / 256, 256, 0, stream>>>(counts, dw, db, out);
}
